// PEER_25220047962611
// MI455X (gfx1250) — matrix operand order repaired and checked by reading, not compiled, not run on hardware
//
#include <hip/hip_runtime.h>
#include <hip/hip_bf16.h>

// PEER (product-key memory) forward for MI455X / gfx1250.
// Pipeline: bf16 convert -> WMMA qproj GEMM -> WMMA sim GEMM (grouped) ->
//           split top-k + pruned cartesian + softmax -> gather/GELU/apply.

#define SEQ        1024
#define NDIM       1024
#define KDIM       512
#define NHEADS     8
#define NKEYS      128
#define TK         16
#define NEXPERTS   16384
#define QFEAT      8192   // KDIM * NHEADS * 2

typedef __bf16 bf16_t;
typedef __attribute__((ext_vector_type(16))) __bf16 v16bf;
typedef __attribute__((ext_vector_type(8)))  __bf16 v8bf;
typedef __attribute__((ext_vector_type(8)))  float  v8f;

__device__ inline unsigned short f2bf_bits(float f) {
  unsigned u = __builtin_bit_cast(unsigned, f);
  unsigned r = u + 0x7FFFu + ((u >> 16) & 1u);   // round to nearest even
  return (unsigned short)(r >> 16);
}
__device__ inline bf16_t f2bf(float f) {
  unsigned short hs = f2bf_bits(f);
  return __builtin_bit_cast(bf16_t, hs);
}

// ---------------------------------------------------------------- convert
// count must be a multiple of 4 (all our tensors are).
__global__ __launch_bounds__(256) void PEER_cvt_bf16(const float* __restrict__ src,
                                                     bf16_t* __restrict__ dst, int count4) {
  int i = blockIdx.x * blockDim.x + threadIdx.x;
  int stride = gridDim.x * blockDim.x;
  for (; i < count4; i += stride) {
    float4 v = ((const float4*)src)[i];
    unsigned lo = (unsigned)f2bf_bits(v.x) | ((unsigned)f2bf_bits(v.y) << 16);
    unsigned hi = (unsigned)f2bf_bits(v.z) | ((unsigned)f2bf_bits(v.w) << 16);
    uint2 pk; pk.x = lo; pk.y = hi;
    ((uint2*)dst)[i] = pk;
  }
}

// ----------------------------------------------------- WMMA NT-GEMM (bf16)
// C[M,N] = A[M,K] * B[N,K]^T, K contiguous in both (NT). Wave computes 64x64.
__device__ inline v16bf load_a_frag(const bf16_t* A, int lda, int row, int k0, int lane) {
  int r  = row + (lane & 15);
  int kb = k0 + ((lane >> 4) << 4);
  const bf16_t* p = A + (size_t)r * lda + kb;
  v8bf lo = *(const v8bf*)(p);
  v8bf hi = *(const v8bf*)(p + 8);
  v16bf a;
#pragma unroll
  for (int i = 0; i < 8; ++i) { a[i] = lo[i]; a[i + 8] = hi[i]; }
  return a;
}
__device__ inline v16bf load_b_frag(const bf16_t* B, int ldb, int col, int k0, int lane) {
  int c  = col + (lane & 15);
  int kb = k0 + ((lane >> 4) << 4);
  return *(const v16bf*)(B + (size_t)c * ldb + kb);
}

__device__ inline void store_c(float* p, float v)  { *p = v; }
__device__ inline void store_c(bf16_t* p, float v) { *p = f2bf(v); }

template <typename CT>
__global__ __launch_bounds__(256) void PEER_wmma_nt_gemm(
    const bf16_t* __restrict__ A, const bf16_t* __restrict__ B, CT* __restrict__ C,
    int K, int lda, int ldb, int ldc,
    int ntm, int ntn, int ngroups, int num_h,
    long long aoff_p, long long aoff_h,
    long long boff_p, long long boff_h,
    long long coff_p, long long coff_h)
{
  int wave = (int)((blockIdx.x * blockDim.x + threadIdx.x) >> 5);
  int lane = threadIdx.x & 31;
  int tiles_per_group = ntm * ntn;
  if (wave >= tiles_per_group * ngroups) return;   // wave-uniform exit

  int g  = wave / tiles_per_group;
  int t  = wave % tiles_per_group;
  int tm = t / ntn, tn = t % ntn;
  int p  = g / num_h, h = g % num_h;

  const bf16_t* Ag = A + aoff_p * p + aoff_h * h;
  const bf16_t* Bg = B + boff_p * p + boff_h * h;
  CT*           Cg = C + coff_p * p + coff_h * h;

  int m0 = tm * 64, n0 = tn * 64;
  v8f acc[4][4];
#pragma unroll
  for (int i = 0; i < 4; ++i)
#pragma unroll
    for (int j = 0; j < 4; ++j) {
      v8f z = {0.f, 0.f, 0.f, 0.f, 0.f, 0.f, 0.f, 0.f};
      acc[i][j] = z;
    }

  for (int k0 = 0; k0 < K; k0 += 32) {
    v16bf a[4], b[4];
#pragma unroll
    for (int i = 0; i < 4; ++i) a[i] = load_a_frag(Ag, lda, m0 + 16 * i, k0, lane);
#pragma unroll
    for (int j = 0; j < 4; ++j) b[j] = load_b_frag(Bg, ldb, n0 + 16 * j, k0, lane);
#pragma unroll
    for (int i = 0; i < 4; ++i)
#pragma unroll
      for (int j = 0; j < 4; ++j)
        acc[i][j] = __builtin_amdgcn_wmma_f32_16x16x32_bf16(
            false, a[i], false, b[j], (short)0, acc[i][j], false, false);
  }

  // C/D layout: element r of v8f in lane L -> M = r + 8*(L>>4), N = L&15
  int cl = lane & 15;
  int mb = (lane >> 4) * 8;
#pragma unroll
  for (int i = 0; i < 4; ++i)
#pragma unroll
    for (int j = 0; j < 4; ++j) {
      CT* cp = Cg + (size_t)(m0 + 16 * i + mb) * ldc + (n0 + 16 * j + cl);
#pragma unroll
      for (int r = 0; r < 8; ++r) store_c(cp + (size_t)r * ldc, acc[i][j][r]);
    }
}

// ------------------------------------------------------------------ top-k
__device__ inline void insert16(float v, int id, float (&s)[TK], int (&ind)[TK]) {
#pragma unroll
  for (int j = 0; j < TK; ++j) {
    bool gt  = v > s[j];
    float ns = gt ? v : s[j];
    float nv = gt ? s[j] : v;
    int nis  = gt ? id : ind[j];
    int niv  = gt ? ind[j] : id;
    s[j] = ns; ind[j] = nis;
    v = nv; id = niv;
  }
}

// Two threads per (n,h): even thread -> p=0 list, odd thread -> p=1 list.
// Odd publishes its sorted list via LDS; even does pruned cartesian + softmax.
// Pruning: a pair (a,b) can be in the top-16 of sx[a]+sy[b] (both sorted desc)
// only if (a+1)*(b+1) <= 16  ->  50 candidates instead of 256.
__global__ __launch_bounds__(256) void PEER_topk(
    const float* __restrict__ sim,   // [2][SEQ][NHEADS][NKEYS]
    int* __restrict__ rows,          // [SEQ*NHEADS*TK]
    float* __restrict__ wts)         // [SEQ*NHEADS*TK]
{
  int gt = blockIdx.x * blockDim.x + threadIdx.x;
  int prob = gt >> 1;                // (n*NHEADS + h)
  int p    = gt & 1;
  if (prob >= SEQ * NHEADS) return;
  int h = prob & 7;

  __shared__ float ls_s[128][TK];    // p=1 lists, one slot per local problem
  __shared__ int   ls_i[128][TK];
  int lp = threadIdx.x >> 1;

  const float* s = sim + (size_t)p * (SEQ * NHEADS * NKEYS) + (size_t)prob * NKEYS;
  float sv[TK]; int si[TK];
#pragma unroll
  for (int i = 0; i < TK; ++i) { sv[i] = -3.4e38f; si[i] = 0; }
  for (int k = 0; k < NKEYS; ++k) insert16(s[k], k, sv, si);

  if (p == 1) {
#pragma unroll
    for (int i = 0; i < TK; ++i) { ls_s[lp][i] = sv[i]; ls_i[lp][i] = si[i]; }
  }
  __syncthreads();

  if (p == 0) {
    float sy[TK]; int iy[TK];
#pragma unroll
    for (int i = 0; i < TK; ++i) { sy[i] = ls_s[lp][i]; iy[i] = ls_i[lp][i]; }

    float fs[TK]; int fi[TK];
#pragma unroll
    for (int i = 0; i < TK; ++i) { fs[i] = -3.4e38f; fi[i] = 0; }
#pragma unroll
    for (int a = 0; a < TK; ++a)
#pragma unroll
      for (int b = 0; b < TK; ++b)
        if ((a + 1) * (b + 1) <= TK)                       // compile-time prune
          insert16(sv[a] + sy[b], si[a] * NKEYS + iy[b], fs, fi);

    float m = fs[0], sum = 0.f, e[TK];
#pragma unroll
    for (int i = 0; i < TK; ++i) { e[i] = __expf(fs[i] - m); sum += e[i]; }
    float inv = 1.f / sum;
    size_t base = (size_t)prob * TK;
#pragma unroll
    for (int i = 0; i < TK; ++i) {
      wts[base + i]  = e[i] * inv;
      rows[base + i] = fi[i] + h * NEXPERTS;
    }
  }
}

// -------------------------------------------------- expert gather / apply
__global__ __launch_bounds__(256) void PEER_experts(
    const float* __restrict__ x,       // [SEQ][NDIM]
    const float* __restrict__ down,    // [NEXPERTS*NHEADS][NDIM]
    const float* __restrict__ up,      // [NEXPERTS*NHEADS][NDIM]
    const int*   __restrict__ rows,
    const float* __restrict__ wts,
    float* __restrict__ out)           // [SEQ][NDIM]
{
  int n = blockIdx.x;
  int tid  = threadIdx.x;              // 256 threads, each owns 4 dims
  int lane = tid & 31, wid = tid >> 5;

  __shared__ int   s_rows[NHEADS * TK];
  __shared__ float s_w[NHEADS * TK];
  __shared__ float s_red[8][TK];
  __shared__ float s_coef[TK];

  if (tid < NHEADS * TK) {             // preload all 128 (row, weight) pairs
    s_rows[tid] = rows[(size_t)n * NHEADS * TK + tid];
    s_w[tid]    = wts [(size_t)n * NHEADS * TK + tid];
  }
  const float4 xv = *(const float4*)(x + (size_t)n * NDIM + tid * 4);
  float a0 = 0.f, a1 = 0.f, a2 = 0.f, a3 = 0.f;
  __syncthreads();

  for (int h = 0; h < NHEADS; ++h) {
    const int* rh = s_rows + h * TK;

    // Hide HBM latency behind the reduction barriers: warm this head's up
    // rows and the next head's down rows (global_prefetch_b8).
#pragma unroll
    for (int k = 0; k < TK; ++k)
      __builtin_prefetch(up + (size_t)rh[k] * NDIM + tid * 4, 0, 1);
    if (h + 1 < NHEADS) {
#pragma unroll
      for (int k = 0; k < TK; ++k)
        __builtin_prefetch(down + (size_t)s_rows[(h + 1) * TK + k] * NDIM + tid * 4, 0, 1);
    }

    float part[TK];
#pragma unroll
    for (int k = 0; k < TK; ++k) {
      const float4 wv = *(const float4*)(down + (size_t)rh[k] * NDIM + tid * 4);
      part[k] = xv.x * wv.x + xv.y * wv.y + xv.z * wv.z + xv.w * wv.w;
    }
#pragma unroll
    for (int k = 0; k < TK; ++k) {
      float v = part[k];
#pragma unroll
      for (int off = 16; off > 0; off >>= 1) v += __shfl_xor(v, off, 32);
      if (lane == 0) s_red[wid][k] = v;
    }
    __syncthreads();
    if (tid < TK) {
      float d = 0.f;
#pragma unroll
      for (int w = 0; w < 8; ++w) d += s_red[w][tid];
      float g = 0.5f * d * (1.0f + erff(d * 0.70710678118654752f));  // exact GELU
      s_coef[tid] = s_w[h * TK + tid] * g;
    }
    __syncthreads();
#pragma unroll
    for (int k = 0; k < TK; ++k) {
      float c = s_coef[k];
      const float4 uv = *(const float4*)(up + (size_t)rh[k] * NDIM + tid * 4);
      a0 += c * uv.x; a1 += c * uv.y; a2 += c * uv.z; a3 += c * uv.w;
    }
    __syncthreads();                   // s_red/s_coef reused next head
  }

  float4 o; o.x = a0; o.y = a1; o.z = a2; o.w = a3;
  *(float4*)(out + (size_t)n * NDIM + tid * 4) = o;
}

// ------------------------------------------------------------------ launch
extern "C" void kernel_launch(void* const* d_in, const int* in_sizes, int n_in,
                              void* d_out, int out_size, void* d_ws, size_t ws_size,
                              hipStream_t stream) {
  const float* x    = (const float*)d_in[0];   // (1,1024,1024)
  const float* Wq   = (const float*)d_in[1];   // (8192,1024)
  const float* keys = (const float*)d_in[2];   // (8,128,2,512)
  const float* up   = (const float*)d_in[3];   // (131072,1024)
  const float* down = (const float*)d_in[4];   // (131072,1024)
  float* out = (float*)d_out;

  char* ws = (char*)d_ws;
  bf16_t* xb    = (bf16_t*)(ws);                         //  2 MB
  bf16_t* Wqb   = (bf16_t*)(ws + (2u   << 20));          // 16 MB
  bf16_t* keysb = (bf16_t*)(ws + (18u  << 20));          //  2 MB
  bf16_t* qb    = (bf16_t*)(ws + (20u  << 20));          // 16 MB
  float*  sim   = (float*) (ws + (36u  << 20));          //  8 MB
  int*    rows  = (int*)   (ws + (44u  << 20));          // 0.5 MB
  float*  wts   = (float*) (ws + (45u  << 20));          // 0.5 MB

  // 1) convert to bf16 (vectorized x4)
  PEER_cvt_bf16<<<256,  256, 0, stream>>>(x,    xb,    SEQ * NDIM / 4);
  PEER_cvt_bf16<<<1024, 256, 0, stream>>>(Wq,   Wqb,   QFEAT * NDIM / 4);
  PEER_cvt_bf16<<<256,  256, 0, stream>>>(keys, keysb, NHEADS * NKEYS * 2 * KDIM / 4);

  // 2) q = x @ Wq^T : M=1024, N=8192, K=1024 (NT, bf16 WMMA), q stored bf16
  {
    int ntm = SEQ / 64, ntn = QFEAT / 64;                // 16 x 128 tiles
    int waves = ntm * ntn;
    int blocks = (waves * 32 + 255) / 256;
    PEER_wmma_nt_gemm<bf16_t><<<blocks, 256, 0, stream>>>(
        xb, Wqb, qb, NDIM, NDIM, NDIM, QFEAT,
        ntm, ntn, /*groups*/1, /*num_h*/1, 0, 0, 0, 0, 0, 0);
  }

  // 3) sim[p,n,h,k] = q[n,(p,h,:)] . keys[h,k,p,:] : 16 grouped NT-GEMMs
  {
    int ntm = SEQ / 64, ntn = NKEYS / 64;                // 16 x 2 tiles
    int waves = ntm * ntn * 16;
    int blocks = (waves * 32 + 255) / 256;
    PEER_wmma_nt_gemm<float><<<blocks, 256, 0, stream>>>(
        qb, keysb, sim, KDIM,
        /*lda*/QFEAT, /*ldb*/2 * KDIM, /*ldc*/NHEADS * NKEYS,
        ntm, ntn, /*groups*/16, /*num_h*/NHEADS,
        /*aoff_p*/ (long long)NHEADS * KDIM, /*aoff_h*/ KDIM,
        /*boff_p*/ KDIM,                     /*boff_h*/ (long long)NKEYS * 2 * KDIM,
        /*coff_p*/ (long long)SEQ * NHEADS * NKEYS, /*coff_h*/ NKEYS);
  }

  // 4) split dual top-16 (2 threads/problem) + pruned cartesian + softmax
  PEER_topk<<<(SEQ * NHEADS * 2 + 255) / 256, 256, 0, stream>>>(sim, rows, wts);

  // 5) gather experts, dot + GELU + weighted sum (deterministic, no atomics)
  PEER_experts<<<SEQ, 256, 0, stream>>>(x, down, up, rows, wts, out);
}
